// mamba_4D_23845658428251
// MI455X (gfx1250) — compile-verified, matrix operand order checked by reading, not run on hardware
//
#include <hip/hip_runtime.h>
#include <hip/hip_bf16.h>

typedef __attribute__((ext_vector_type(16))) _Float16 v16h;
typedef __attribute__((ext_vector_type(8)))  _Float16 v8h;
typedef __attribute__((ext_vector_type(8)))  float    v8f;

// ---------------- problem constants ----------------
constexpr int Bb = 4, Hh = 64, Ww = 64;
constexpr int Dm = 192, Di = 384;
constexpr int Rr = 12, Nn = 16, Kk = 4, Pp = 32;
constexpr int Ll = Hh * Ww;             // 4096
constexpr int CSc = 64, NCh = Ll / CSc; // 64 chunks
constexpr int Gg = Bb * Ll;             // 16384 rows
constexpr int CPAD = 48;                // 44 -> 48 padded proj cols

// ---------------- helpers ----------------
__device__ __forceinline__ float gelu_exact(float v) {
    return 0.5f * v * (1.0f + erff(v * 0.70710678118654752f));
}
__device__ __forceinline__ float softplus_f(float v) {
    return (v > 20.0f) ? v : logf(1.0f + expf(v));
}
// cross-scan index map (self-inverse for every direction; H=W=64)
__device__ __forceinline__ int rowmap(int k, int q) {
    if (k == 0) return q;
    if (k == 1) { int w = q >> 6, h = q & 63; return (h << 6) | w; }
    if (k == 2) return (Ll - 1) - q;
    int q3 = (Ll - 1) - q; int w = q3 >> 6, h = q3 & 63; return (h << 6) | w;
}

// Fragment-layout LDS tiles: a 16x32 f16 tile is stored as [lane][16] halves so
// each lane reads its whole WMMA operand slice as one contiguous 32B v16h.
// Inverse of ISA 7.12.2 16-bit A/B layout:
//   lane = m + 16*bit3(k),  e = (k&7) + 8*(k>=16)
__device__ __forceinline__ int frag_a_off(int m, int k) {
    return ((m + ((k >> 3 & 1) << 4)) << 4) + (k & 7) + ((k >> 4) << 3);
}
__device__ __forceinline__ int frag_b_off(int k, int n) {
    return ((n + ((k >> 3 & 1) << 4)) << 4) + (k & 7) + ((k >> 4) << 3);
}
__device__ __forceinline__ v16h load_frag(const v16h* tile) {
    return tile[threadIdx.x & 31];
}
__device__ __forceinline__ v8f wmma_f16(v16h a, v16h b, v8f c) {
    return __builtin_amdgcn_wmma_f32_16x16x32_f16(false, a, false, b, (short)0, c, false, false);
}

// ---------------- staging kernels ----------------
__global__ void cvt_f16_k(const float* __restrict__ s, _Float16* __restrict__ d, int n) {
    int i = blockIdx.x * blockDim.x + threadIdx.x;
    if (i < n) d[i] = (_Float16)s[i];
}
__global__ void pad_xpw_k(const float* __restrict__ s, _Float16* __restrict__ d) {
    int i = blockIdx.x * blockDim.x + threadIdx.x;
    if (i >= Kk * CPAD * Di) return;
    int dd = i % Di; int c = (i / Di) % CPAD; int k = i / (Di * CPAD);
    d[i] = (c < Rr + 2 * Nn) ? (_Float16)s[(k * (Rr + 2 * Nn) + c) * Di + dd] : (_Float16)0.0f;
}
__global__ void zero_k(float* __restrict__ p, int n) {
    int i = blockIdx.x * blockDim.x + threadIdx.x;
    if (i < n) p[i] = 0.0f;
}

// ---------------- K1: in-projection GEMM (WMMA) ----------------
__global__ void gemm_inproj_k(const _Float16* __restrict__ X, const _Float16* __restrict__ Wt,
                              float* __restrict__ xi_raw, float* __restrict__ zbuf) {
    __shared__ v16h AsT[4][32];   // 4 row tiles
    __shared__ v16h BsT[4][32];   // 4 col tiles
    int rowBase = blockIdx.x * 64, colBase = blockIdx.y * 64;
    int tid = threadIdx.x, wave = tid >> 5, lane = tid & 31;
    int rt = wave & 3, cp = wave >> 2;
    v8f acc0 = {}, acc1 = {};
    for (int kk = 0; kk < Dm / 32; ++kk) {
        int kb = kk * 32;
        // 256 jobs: (row-or-col index, k-group of 8)
        int m64 = tid >> 2, kg = (tid & 3) * 8;
        v8h va = *(const v8h*)(X + (rowBase + m64) * Dm + kb + kg);
        *(v8h*)((_Float16*)&AsT[m64 >> 4][0] + frag_a_off(m64 & 15, kg)) = va;
        v8h vb = *(const v8h*)(Wt + (colBase + m64) * Dm + kb + kg);
        *(v8h*)((_Float16*)&BsT[m64 >> 4][0] + frag_b_off(kg, m64 & 15)) = vb;
        __syncthreads();
        v16h a  = load_frag(&AsT[rt][0]);
        v16h b0 = load_frag(&BsT[cp * 2 + 0][0]);
        v16h b1 = load_frag(&BsT[cp * 2 + 1][0]);
        acc0 = wmma_f16(a, b0, acc0);
        acc1 = wmma_f16(a, b1, acc1);
        __syncthreads();
    }
#pragma unroll
    for (int jj = 0; jj < 8; ++jj) {
        int row = rowBase + rt * 16 + jj + ((lane >> 4) << 3);
#pragma unroll
        for (int cc = 0; cc < 2; ++cc) {
            int col = colBase + (cp * 2 + cc) * 16 + (lane & 15);
            float v = (cc == 0) ? acc0[jj] : acc1[jj];
            if (col < Di) xi_raw[row * Di + col] = v;
            else          zbuf[row * Di + (col - Di)] = gelu_exact(v);
        }
    }
}

// ---------------- K2: depthwise 3x3 conv + gelu ----------------
__global__ void conv_gelu_k(const float* __restrict__ xi_raw, const float* __restrict__ cw,
                            const float* __restrict__ cb, float* __restrict__ xi32,
                            _Float16* __restrict__ xi16) {
    int i = blockIdx.x * blockDim.x + threadIdx.x;
    if (i >= Gg * Di) return;
    int c = i % Di; int w = (i / Di) % Ww; int h = (i / Di / Ww) % Hh; int b = i / (Di * Ww * Hh);
    float acc = cb[c];
#pragma unroll
    for (int dy = -1; dy <= 1; ++dy) {
        int hh = h + dy; if (hh < 0 || hh >= Hh) continue;
#pragma unroll
        for (int dx = -1; dx <= 1; ++dx) {
            int ww = w + dx; if (ww < 0 || ww >= Ww) continue;
            acc += xi_raw[((b * Hh + hh) * Ww + ww) * Di + c] * cw[((dy + 1) * 3 + (dx + 1)) * Di + c];
        }
    }
    float g = gelu_exact(acc);
    xi32[i] = g;
    xi16[i] = (_Float16)g;
}

// ---------------- K3: per-direction projection GEMM (WMMA, gathered rows) ----------------
__global__ void gemm_xdbl_k(const _Float16* __restrict__ Xi, const _Float16* __restrict__ Wp,
                            float* __restrict__ xdbl) {
    __shared__ v16h AsT[8][32];   // 8 row tiles (128 rows)
    __shared__ v16h BsT[3][32];   // 3 col tiles (48 cols)
    int rowBase = blockIdx.x * 128;
    int kdir = blockIdx.y;
    int tid = threadIdx.x, wave = tid >> 5, lane = tid & 31;
    v8f acc[3] = {v8f{}, v8f{}, v8f{}};
    for (int kk = 0; kk < Di / 32; ++kk) {
        int kb = kk * 32;
        // A: 128 rows x 4 k-groups = 512 jobs
        for (int i = tid; i < 512; i += 256) {
            int m128 = i >> 2, kg = (i & 3) * 8;
            int g = rowBase + m128; int b = g >> 12; int q = g & (Ll - 1);
            v8h v = *(const v8h*)(Xi + ((b << 12) | rowmap(kdir, q)) * Di + kb + kg);
            *(v8h*)((_Float16*)&AsT[m128 >> 4][0] + frag_a_off(m128 & 15, kg)) = v;
        }
        // B: 48 cols x 4 k-groups = 192 jobs
        if (tid < 192) {
            int c = tid >> 2, kg = (tid & 3) * 8;
            v8h v = *(const v8h*)(Wp + (kdir * CPAD + c) * Di + kb + kg);
            *(v8h*)((_Float16*)&BsT[c >> 4][0] + frag_b_off(kg, c & 15)) = v;
        }
        __syncthreads();
        v16h a = load_frag(&AsT[wave][0]);
#pragma unroll
        for (int ct = 0; ct < 3; ++ct) {
            v16h b = load_frag(&BsT[ct][0]);
            acc[ct] = wmma_f16(a, b, acc[ct]);
        }
        __syncthreads();
    }
#pragma unroll
    for (int ct = 0; ct < 3; ++ct) {
#pragma unroll
        for (int jj = 0; jj < 8; ++jj) {
            int g = rowBase + wave * 16 + jj + ((lane >> 4) << 3);
            int c = ct * 16 + (lane & 15);
            xdbl[(g * Kk + kdir) * CPAD + c] = acc[ct][jj];
        }
    }
}

// ---------------- K4: dt = softplus(dts + bias) ----------------
__global__ void dt_softplus_k(const float* __restrict__ xdbl, const float* __restrict__ dt_bias,
                              float* __restrict__ dtb) {
    int i = blockIdx.x * blockDim.x + threadIdx.x;
    if (i >= Gg * Kk * Rr) return;
    int r = i % Rr; int k = (i / Rr) % Kk; int g = i / (Rr * Kk);
    dtb[i] = softplus_f(xdbl[(g * Kk + k) * CPAD + r] + dt_bias[k * Rr + r]);
}

// ---------------- K5: per-chunk cumsum of dt*A and chunk decay ----------------
__global__ void chunk_cum_k(const float* __restrict__ dtb, const float* __restrict__ A_logs,
                            float* __restrict__ cumb, float* __restrict__ cdec) {
    int i = blockIdx.x * blockDim.x + threadIdx.x;
    if (i >= Bb * Kk * NCh * Rr) return;
    int r = i % Rr; int z = (i / Rr) % NCh; int k = (i / (Rr * NCh)) % Kk; int b = i / (Rr * NCh * Kk);
    float A = -expf(A_logs[k * Rr + r]);
    float acc = 0.0f;
    for (int t = 0; t < CSc; ++t) {
        int g = b * Ll + z * CSc + t;
        acc += dtb[(g * Kk + k) * Rr + r] * A;
        cumb[(g * Kk + k) * Rr + r] = acc;
    }
    cdec[((b * NCh + z) * Kk + k) * Rr + r] = expf(acc);
}

// ---------------- K6: chunk states GEMM (WMMA): states = B^T @ (w*x) ----------------
__global__ void states_k(const float* __restrict__ xdbl, const _Float16* __restrict__ Xi,
                         const float* __restrict__ cumb, const float* __restrict__ dtb,
                         float* __restrict__ states) {
    __shared__ v16h  BcTT[2][32];          // B^T tiles: [ks]
    __shared__ float wlds[64][Rr];
    __shared__ v16h  wxT[Rr][4][32];       // (w*x) tiles: [r][ks*2+ct]
    int z = blockIdx.x, k = blockIdx.y, b = blockIdx.z;
    int tid = threadIdx.x, wave = tid >> 5, lane = tid & 31;
    // B^T: 16 n-rows x 8 s-groups = 128 jobs (gathered f32 -> f16 v8h)
    if (tid < 128) {
        int n = tid >> 3, s0 = (tid & 7) * 8;
        v8h v;
#pragma unroll
        for (int e = 0; e < 8; ++e) {
            int g = b * Ll + z * CSc + s0 + e;
            v[e] = (_Float16)xdbl[(g * Kk + k) * CPAD + Rr + n];
        }
        *(v8h*)((_Float16*)&BcTT[s0 >> 5][0] + frag_a_off(n, s0 & 31)) = v;
    }
    for (int i = tid; i < 64 * Rr; i += 256) {
        int r = i % Rr, s = i / Rr;
        int g = b * Ll + z * CSc + s;
        float cl = cumb[((b * Ll + z * CSc + 63) * Kk + k) * Rr + r];
        wlds[s][r] = expf(cl - cumb[(g * Kk + k) * Rr + r]) * dtb[(g * Kk + k) * Rr + r];
    }
    __syncthreads();
    // wx: 12r x 8 s-groups x 32p = 3072 jobs
    for (int i = tid; i < Rr * 8 * 32; i += 256) {
        int p = i & 31, sg = (i >> 5) & 7, r = i >> 8;
        int s0 = sg * 8;
        v8h v;
#pragma unroll
        for (int e = 0; e < 8; ++e) {
            int s = s0 + e, q = z * CSc + s;
            float xv = (float)Xi[((b << 12) | rowmap(k, q)) * Di + r * 32 + p];
            v[e] = (_Float16)(wlds[s][r] * xv);
        }
        *(v8h*)((_Float16*)&wxT[r][(s0 >> 5) * 2 + (p >> 4)][0] + frag_b_off(s0 & 31, p & 15)) = v;
    }
    __syncthreads();
    for (int job = wave * 3; job < wave * 3 + 3; ++job) {   // 24 jobs: (r, col-tile)
        int r = job >> 1, ct = job & 1;
        v8f acc = {};
#pragma unroll
        for (int ks = 0; ks < 2; ++ks) {
            v16h a  = load_frag(&BcTT[ks][0]);
            v16h bf = load_frag(&wxT[r][ks * 2 + ct][0]);
            acc = wmma_f16(a, bf, acc);
        }
        size_t base = ((size_t)((b * NCh + z) * Kk + k) * Rr + r) * (Pp * Nn);
#pragma unroll
        for (int jj = 0; jj < 8; ++jj) {
            int n = jj + ((lane >> 4) << 3);
            int p = ct * 16 + (lane & 15);
            states[base + p * Nn + n] = acc[jj];
        }
    }
}

// ---------------- K7: sequential inter-chunk recurrence (in place: states -> prev) ----------------
__global__ void scan_k(float* __restrict__ states, const float* __restrict__ cdec) {
    int i = blockIdx.x * blockDim.x + threadIdx.x;
    if (i >= Bb * Kk * Rr * Pp * Nn) return;
    int n = i % Nn; int p = (i / Nn) % Pp; int r = (i / (Nn * Pp)) % Rr;
    int k = (i / (Nn * Pp * Rr)) % Kk; int b = i / (Nn * Pp * Rr * Kk);
    float carry = 0.0f;
    for (int z = 0; z < NCh; ++z) {
        size_t off = ((size_t)((b * NCh + z) * Kk + k) * Rr + r) * (Pp * Nn) + p * Nn + n;
        float tmp = states[off];
        states[off] = carry;
        carry = carry * cdec[((b * NCh + z) * Kk + k) * Rr + r] + tmp;
    }
}

// ---------------- K8: main chunk kernel (WMMA, fused K=96 GEMM) ----------------
__global__ void chunk_main_k(const float* __restrict__ xdbl, const _Float16* __restrict__ Xi,
                             const float* __restrict__ cumb, const float* __restrict__ dtb,
                             const float* __restrict__ prev, const float* __restrict__ Ds,
                             float* __restrict__ ymerge) {
    __shared__ float    CBs[64][64];
    __shared__ float    CcS[64][16];
    __shared__ float    BcF[64][16];
    __shared__ float    cumS[64][Rr];
    __shared__ float    dtS[64][Rr];
    __shared__ _Float16 prevS[Rr][16 * 32];
    __shared__ v16h     ApT[12][32];      // [rt*3+ks]
    __shared__ v16h     BpT[6][32];       // [ks*2+ct]
    int z = blockIdx.x, k = blockIdx.y, b = blockIdx.z;
    int tid = threadIdx.x, wave = tid >> 5, lane = tid & 31;
    int rt = wave & 3, ct = wave >> 2;
    for (int i = tid; i < 64 * 16; i += 256) {
        int n = i & 15, t = i >> 4;
        int g = b * Ll + z * CSc + t;
        CcS[t][n] = xdbl[(g * Kk + k) * CPAD + Rr + Nn + n];
        BcF[t][n] = xdbl[(g * Kk + k) * CPAD + Rr + n];
    }
    for (int i = tid; i < 64 * Rr; i += 256) {
        int r = i % Rr, t = i / Rr;
        int g = b * Ll + z * CSc + t;
        cumS[t][r] = cumb[(g * Kk + k) * Rr + r];
        dtS[t][r]  = dtb[(g * Kk + k) * Rr + r];
    }
    for (int i = tid; i < Rr * 16 * 32; i += 256) {
        int p = i & 31, n = (i >> 5) & 15, r = i >> 9;
        size_t base = ((size_t)((b * NCh + z) * Kk + k) * Rr + r) * (Pp * Nn);
        prevS[r][n * 32 + p] = (_Float16)prev[base + p * Nn + n];
    }
    __syncthreads();
    for (int i = tid; i < 64 * 64; i += 256) {       // CB = Cc @ Bc^T
        int s = i & 63, t = i >> 6;
        float acc = 0.0f;
#pragma unroll
        for (int n = 0; n < 16; ++n) acc += CcS[t][n] * BcF[s][n];
        CBs[t][s] = acc;
    }
    __syncthreads();
    for (int r = 0; r < Rr; ++r) {
        // A' = [ M_r (64x64) | Cc*exp(cum_r) (64x16) | 0 (64x16) ] : 768 v8h jobs
        for (int i = tid; i < 768; i += 256) {
            int t = i / 12, cg = (i % 12) * 8;
            v8h v;
#pragma unroll
            for (int e = 0; e < 8; ++e) {
                int c = cg + e;
                float f = 0.0f;
                if (c < 64) {
                    if (c <= t) f = CBs[t][c] * expf(cumS[t][r] - cumS[c][r]) * dtS[c][r];
                } else if (c < 80) {
                    f = CcS[t][c - 64] * expf(cumS[t][r]);
                }
                v[e] = (_Float16)f;
            }
            *(v8h*)((_Float16*)&ApT[(t >> 4) * 3 + (cg >> 5)][0] + frag_a_off(t & 15, cg & 31)) = v;
        }
        // B' = [ x_r (64x32) ; prev_r (16x32) ; 0 (16x32) ] : 384 v8h jobs
        for (int i = tid; i < 384; i += 256) {
            int p = i & 31, kg = (i >> 5) * 8;
            v8h v;
#pragma unroll
            for (int e = 0; e < 8; ++e) {
                int kkr = kg + e;
                _Float16 f = (_Float16)0.0f;
                if (kkr < 64) {
                    int q = z * CSc + kkr;
                    f = Xi[((b << 12) | rowmap(k, q)) * Di + r * 32 + p];
                } else if (kkr < 80) {
                    f = prevS[r][(kkr - 64) * 32 + p];
                }
                v[e] = f;
            }
            *(v8h*)((_Float16*)&BpT[(kg >> 5) * 2 + (p >> 4)][0] + frag_b_off(kg & 31, p & 15)) = v;
        }
        __syncthreads();
        v8f acc = {};
#pragma unroll
        for (int ks = 0; ks < 3; ++ks) {
            v16h a  = load_frag(&ApT[rt * 3 + ks][0]);
            v16h bf = load_frag(&BpT[ks * 2 + ct][0]);
            acc = wmma_f16(a, bf, acc);
        }
#pragma unroll
        for (int jj = 0; jj < 8; ++jj) {
            int t = rt * 16 + jj + ((lane >> 4) << 3);
            int p = ct * 16 + (lane & 15);
            float xv = (float)*((const _Float16*)&BpT[(t >> 5) * 2 + ct][0] + frag_b_off(t & 31, p & 15));
            float val = acc[jj] + Ds[(k * Rr + r) * Pp + p] * xv;
            int q = z * CSc + t;
            int l = rowmap(k, q);                      // cross-merge scatter target
            atomicAdd(&ymerge[((b << 12) | l) * Di + r * 32 + p], val);
        }
        __syncthreads();
    }
}

// ---------------- K9: layernorm * gate -> f16 hidden ----------------
__global__ void ln_mul_k(const float* __restrict__ ym, const float* __restrict__ lw,
                         const float* __restrict__ lb, const float* __restrict__ zb,
                         _Float16* __restrict__ hid) {
    __shared__ float red[256];
    int g = blockIdx.x, tid = threadIdx.x;
    float s = 0.0f;
    for (int c = tid; c < Di; c += 256) s += ym[g * Di + c];
    red[tid] = s; __syncthreads();
    for (int o = 128; o > 0; o >>= 1) { if (tid < o) red[tid] += red[tid + o]; __syncthreads(); }
    float mu = red[0] / (float)Di; __syncthreads();
    float v = 0.0f;
    for (int c = tid; c < Di; c += 256) { float d = ym[g * Di + c] - mu; v += d * d; }
    red[tid] = v; __syncthreads();
    for (int o = 128; o > 0; o >>= 1) { if (tid < o) red[tid] += red[tid + o]; __syncthreads(); }
    float rstd = rsqrtf(red[0] / (float)Di + 1e-5f);
    for (int c = tid; c < Di; c += 256) {
        float h = (ym[g * Di + c] - mu) * rstd * lw[c] + lb[c];
        hid[g * Di + c] = (_Float16)(h * zb[g * Di + c]);
    }
}

// ---------------- K10: output projection GEMM (WMMA) ----------------
__global__ void gemm_out_k(const _Float16* __restrict__ Hd, const _Float16* __restrict__ Wo,
                           float* __restrict__ out) {
    __shared__ v16h AsT[4][32];
    __shared__ v16h BsT[4][32];
    int rowBase = blockIdx.x * 64, colBase = blockIdx.y * 64;
    int tid = threadIdx.x, wave = tid >> 5, lane = tid & 31;
    int rt = wave & 3, cp = wave >> 2;
    v8f acc0 = {}, acc1 = {};
    for (int kk = 0; kk < Di / 32; ++kk) {
        int kb = kk * 32;
        int m64 = tid >> 2, kg = (tid & 3) * 8;
        v8h va = *(const v8h*)(Hd + (rowBase + m64) * Di + kb + kg);
        *(v8h*)((_Float16*)&AsT[m64 >> 4][0] + frag_a_off(m64 & 15, kg)) = va;
        v8h vb = *(const v8h*)(Wo + (colBase + m64) * Di + kb + kg);
        *(v8h*)((_Float16*)&BsT[m64 >> 4][0] + frag_b_off(kg, m64 & 15)) = vb;
        __syncthreads();
        v16h a  = load_frag(&AsT[rt][0]);
        v16h b0 = load_frag(&BsT[cp * 2 + 0][0]);
        v16h b1 = load_frag(&BsT[cp * 2 + 1][0]);
        acc0 = wmma_f16(a, b0, acc0);
        acc1 = wmma_f16(a, b1, acc1);
        __syncthreads();
    }
#pragma unroll
    for (int jj = 0; jj < 8; ++jj) {
        int row = rowBase + rt * 16 + jj + ((lane >> 4) << 3);
#pragma unroll
        for (int cc = 0; cc < 2; ++cc) {
            int col = colBase + (cp * 2 + cc) * 16 + (lane & 15);
            out[row * Dm + col] = (cc == 0) ? acc0[jj] : acc1[jj];
        }
    }
}

// ---------------- host ----------------
extern "C" void kernel_launch(void* const* d_in, const int* in_sizes, int n_in,
                              void* d_out, int out_size, void* d_ws, size_t ws_size,
                              hipStream_t stream) {
    (void)in_sizes; (void)n_in; (void)out_size; (void)ws_size;
    const float* x        = (const float*)d_in[0];
    const float* w_in     = (const float*)d_in[1];
    const float* conv_w   = (const float*)d_in[2];
    const float* conv_b   = (const float*)d_in[3];
    const float* x_projw  = (const float*)d_in[4];
    const float* A_logs   = (const float*)d_in[5];
    const float* Ds       = (const float*)d_in[6];
    const float* dt_bias  = (const float*)d_in[7];
    const float* ln_w     = (const float*)d_in[8];
    const float* ln_b     = (const float*)d_in[9];
    const float* w_out    = (const float*)d_in[10];
    float* out = (float*)d_out;

    char* ws = (char*)d_ws;
    size_t off = 0;
    auto alloc = [&](size_t bytes) { void* p = ws + off; off += (bytes + 255) & ~(size_t)255; return p; };

    _Float16* x_f16   = (_Float16*)alloc((size_t)Gg * Dm * 2);
    _Float16* win_f16 = (_Float16*)alloc((size_t)2 * Di * Dm * 2);
    _Float16* wout_f16= (_Float16*)alloc((size_t)Dm * Di * 2);
    _Float16* xpw_f16 = (_Float16*)alloc((size_t)Kk * CPAD * Di * 2);
    float*    xi_raw  = (float*)alloc((size_t)Gg * Di * 4);
    float*    zbuf    = (float*)alloc((size_t)Gg * Di * 4);
    float*    xi32    = (float*)alloc((size_t)Gg * Di * 4);
    _Float16* xi16    = (_Float16*)alloc((size_t)Gg * Di * 2);
    float*    xdbl    = (float*)alloc((size_t)Gg * Kk * CPAD * 4);
    float*    dtb     = (float*)alloc((size_t)Gg * Kk * Rr * 4);
    float*    cumb    = (float*)alloc((size_t)Gg * Kk * Rr * 4);
    float*    cdec    = (float*)alloc((size_t)Bb * NCh * Kk * Rr * 4);
    float*    states  = (float*)alloc((size_t)Bb * NCh * Kk * Rr * Pp * Nn * 4);
    float*    ymerge  = (float*)alloc((size_t)Gg * Di * 4);
    _Float16* hid16   = (_Float16*)alloc((size_t)Gg * Di * 2);

    auto blks = [](int n) { return (n + 255) / 256; };

    cvt_f16_k<<<blks(Gg * Dm), 256, 0, stream>>>(x, x_f16, Gg * Dm);
    cvt_f16_k<<<blks(2 * Di * Dm), 256, 0, stream>>>(w_in, win_f16, 2 * Di * Dm);
    cvt_f16_k<<<blks(Dm * Di), 256, 0, stream>>>(w_out, wout_f16, Dm * Di);
    pad_xpw_k<<<blks(Kk * CPAD * Di), 256, 0, stream>>>(x_projw, xpw_f16);

    gemm_inproj_k<<<dim3(Gg / 64, (2 * Di) / 64), 256, 0, stream>>>(x_f16, win_f16, xi_raw, zbuf);
    conv_gelu_k<<<blks(Gg * Di), 256, 0, stream>>>(xi_raw, conv_w, conv_b, xi32, xi16);
    gemm_xdbl_k<<<dim3(Gg / 128, Kk), 256, 0, stream>>>(xi16, xpw_f16, xdbl);
    dt_softplus_k<<<blks(Gg * Kk * Rr), 256, 0, stream>>>(xdbl, dt_bias, dtb);
    chunk_cum_k<<<blks(Bb * Kk * NCh * Rr), 256, 0, stream>>>(dtb, A_logs, cumb, cdec);
    states_k<<<dim3(NCh, Kk, Bb), 256, 0, stream>>>(xdbl, xi16, cumb, dtb, states);
    scan_k<<<blks(Bb * Kk * Rr * Pp * Nn), 256, 0, stream>>>(states, cdec);
    zero_k<<<blks(Gg * Di), 256, 0, stream>>>(ymerge, Gg * Di);
    chunk_main_k<<<dim3(NCh, Kk, Bb), 256, 0, stream>>>(xdbl, xi16, cumb, dtb, states, Ds, ymerge);
    ln_mul_k<<<Gg, 256, 0, stream>>>(ymerge, ln_w, ln_b, zbuf, hid16);
    gemm_out_k<<<dim3(Gg / 64, Dm / 64), 256, 0, stream>>>(hid16, wout_f16, out);
}
